// EuclideanWorldModel_41704132444514
// MI455X (gfx1250) — matrix operand order repaired and checked by reading, not compiled, not run on hardware
//
#include <hip/hip_runtime.h>

// ---------------------------------------------------------------------------
// MI455X / gfx1250 implementation of the recurrent VAE world model.
//  * All GEMMs run through v_wmma_f32_16x16x32_bf16 (bf16 in, f32 acc).
//  * Tile staging uses GLOBAL_LOAD_ASYNC_TO_LDS_B128 (ASYNCcnt) + barrier.
//  * Encoder hoisted out of the time loop (parallel over B*T tokens).
//  * Per-step weights (~7MB bf16) stay L2-resident across 128 steps.
// ---------------------------------------------------------------------------

typedef __bf16 bf16;
typedef __attribute__((ext_vector_type(16))) __bf16 v16bf;
typedef __attribute__((ext_vector_type(8)))  __bf16 v8bf;
typedef __attribute__((ext_vector_type(8)))  float  v8f;

#define B_   1024
#define T_   128
#define OBS_ 256
#define LAT_ 64
#define HID_ 512
#define ENC_ 512
#define GP_  128          // 2*LAT
#define BT_  (B_ * T_)

#define BM 64
#define BN 64
#define BK 32
#define BKP 48            // padded LDS row stride (keeps 16/32B alignment, dodges bank conflicts)

// ---------------------------------------------------------------------------
// Tiled WMMA GEMM: C[M,N] = act( A[M,K] * W[N,K]^T + bias ), optional accumulate
// Grid: (N/64, M/64), 128 threads (4 waves). Each wave owns a 32x32 sub-tile
// as a 2x2 grid of 16x16 WMMA accumulators.
// ACT: 0 = linear, 1 = ELU.  ACCUM: 1 = add pre-existing Cf before bias/act.
// ---------------------------------------------------------------------------
template <int ACT, int ACCUM>
__global__ __launch_bounds__(128)
void gemm_bf16_wmma(const bf16* __restrict__ A, int lda,
                    const bf16* __restrict__ W, int ldw,
                    const float* __restrict__ bias,
                    float* __restrict__ Cf, int ldc,
                    bf16* __restrict__ Cb, int ldcb,
                    int K)
{
    __shared__ __align__(64) bf16 As[BM * BKP];
    __shared__ __align__(64) bf16 Bs[BN * BKP];

    const int tid  = threadIdx.x;
    const int wave = tid >> 5;
    const int lane = tid & 31;
    const long bm  = (long)blockIdx.y * BM;
    const long bn  = (long)blockIdx.x * BN;
    const int wm   = (wave >> 1) * 32;   // wave M offset inside block tile
    const int wn   = (wave & 1) * 32;    // wave N offset inside block tile
    const int lr   = tid >> 1;           // 0..63: tile row loaded by this thread
    const int lc   = (tid & 1) * 16;     // 0 or 16: K-column chunk (16 bf16 = 32B)
    const int hk   = lane >> 4;          // lane half (K-group select per ISA layout)
    const int lm   = lane & 15;          // M/N index within 16

    v8f acc[2][2] = {};

    // Async global->LDS staging (GVS mode): mem = SGPR_base + VGPR_i32_offset.
    const long a_row = (bm + lr) * (long)lda + lc;     // element offset of this thread's A chunk
    const long w_row = (bn + lr) * (long)ldw + lc;
    const unsigned long long Abase = (unsigned long long)A;
    const unsigned long long Wbase = (unsigned long long)W;
    const unsigned lds_a = (unsigned)(unsigned long long)&As[lr * BKP + lc];
    const unsigned lds_b = (unsigned)(unsigned long long)&Bs[lr * BKP + lc];
    const bf16* Apf = A + a_row;   // prefetch pointers (next K tile)
    const bf16* Wpf = W + w_row;

    for (int k0 = 0; k0 < K; k0 += BK) {
        const unsigned ago = (unsigned)((a_row + k0) * 2);   // byte offsets (fit in i32)
        const unsigned wgo = (unsigned)((w_row + k0) * 2);
        // 16B per lane straight into LDS, tracked by ASYNCcnt (no VGPR staging).
        asm volatile("global_load_async_to_lds_b128 %0, %1, %2"
                     :: "v"(lds_a), "v"(ago), "s"(Abase) : "memory");
        asm volatile("global_load_async_to_lds_b128 %0, %1, %2"
                     :: "v"(lds_b), "v"(wgo), "s"(Wbase) : "memory");
        if (k0 + BK < K) {   // gfx1250 global_prefetch_b8 on next K tile (near scope)
            __builtin_prefetch(Apf + k0 + BK, 0, 3);
            __builtin_prefetch(Wpf + k0 + BK, 0, 3);
        }
        asm volatile("s_wait_asynccnt 0x0" ::: "memory");
        __syncthreads();

        // A fragment (16x32 bf16): lane lm = row M; lanes 0-15 K=0..15,
        // lanes 16-31 K=16..31 -> two 8-element contiguous LDS loads.
        v16bf afrag[2];
#pragma unroll
        for (int mi = 0; mi < 2; ++mi) {
            const bf16* p = &As[(wm + mi * 16 + lm) * BKP + hk * 16];
            v8bf lo = *(const v8bf*)p;
            v8bf hi = *(const v8bf*)(p + 8);
            v16bf f;
#pragma unroll
            for (int j = 0; j < 8; ++j) { f[j] = lo[j]; f[j + 8] = hi[j]; }
            afrag[mi] = f;
        }
        // B fragment (32x16): lane lm = col N; lanes 0-15 K=0..15,
        // lanes 16-31 K=16..31 -> one 16-element contiguous LDS load.
        v16bf bfrag[2];
#pragma unroll
        for (int ni = 0; ni < 2; ++ni)
            bfrag[ni] = *(const v16bf*)(&Bs[(wn + ni * 16 + lm) * BKP + hk * 16]);

#pragma unroll
        for (int mi = 0; mi < 2; ++mi)
#pragma unroll
            for (int ni = 0; ni < 2; ++ni)
                acc[mi][ni] = __builtin_amdgcn_wmma_f32_16x16x32_bf16(
                    false, afrag[mi], false, bfrag[ni],
                    (short)0, acc[mi][ni], false, false);
        __syncthreads();   // all waves done reading before next async writes
    }

    // Epilogue: C/D layout — VGPR v holds row M = v + 8*hk, col N = lane&15.
#pragma unroll
    for (int ni = 0; ni < 2; ++ni) {
        const long col = bn + wn + ni * 16 + lm;
        const float bv = bias ? bias[col] : 0.0f;
#pragma unroll
        for (int mi = 0; mi < 2; ++mi) {
#pragma unroll
            for (int v = 0; v < 8; ++v) {
                const long row = bm + wm + mi * 16 + hk * 8 + v;
                float x = acc[mi][ni][v];
                if (ACCUM) x += Cf[row * (long)ldc + col];
                x += bv;
                if (ACT == 1) x = (x > 0.0f) ? x : (__expf(x) - 1.0f);
                if (Cf) Cf[row * (long)ldc + col] = x;
                if (Cb) Cb[row * (long)ldcb + col] = (bf16)x;
            }
        }
    }
}

// ---------------------------------------------------------------------------
// Elementwise kernels
// ---------------------------------------------------------------------------
__global__ __launch_bounds__(256)
void f32_to_bf16_kernel(const float* __restrict__ in, bf16* __restrict__ out, long n)
{
    long i = (long)blockIdx.x * 256 + threadIdx.x;
    if (i < n) out[i] = (bf16)in[i];
}

__global__ __launch_bounds__(256)
void init_state_kernel(float* __restrict__ hf, bf16* __restrict__ hb, long n)
{
    long i = (long)blockIdx.x * 256 + threadIdx.x;
    if (i < n) { hf[i] = 0.0f; hb[i] = (bf16)0.0f; }
}

// z = mu_q + exp(ls_q)*eps ; mu out ; per-row KL reduction. Grid: B blocks x 64 threads.
__global__ __launch_bounds__(64)
void latent_step_kernel(const float* __restrict__ pri, const float* __restrict__ pos,
                        const float* __restrict__ eps, bf16* __restrict__ zb,
                        float* __restrict__ mu_out, float* __restrict__ kl_out, int t)
{
    const int b = blockIdx.x;
    const int j = threadIdx.x;
    const float mu_p = pri[b * GP_ + j];
    const float ls_p = pri[b * GP_ + LAT_ + j];
    const float mu_q = pos[b * GP_ + j];
    const float ls_q = pos[b * GP_ + LAT_ + j];
    const float e    = eps[((long)b * T_ + t) * LAT_ + j];
    const float zv   = mu_q + __expf(ls_q) * e;
    zb[b * LAT_ + j] = (bf16)zv;
    mu_out[((long)b * T_ + t) * LAT_ + j] = mu_q;

    const float d = mu_q - mu_p;
    float term = ls_p - ls_q
               + (__expf(2.0f * ls_q) + d * d) / (2.0f * __expf(2.0f * ls_p))
               - 0.5f;
    __shared__ float red[64];
    red[j] = term;
    __syncthreads();
#pragma unroll
    for (int s = 32; s > 0; s >>= 1) {
        if (j < s) red[j] += red[j + s];
        __syncthreads();
    }
    if (j == 0) kl_out[(long)b * T_ + t] = red[0];
}

// GRU gate fusion: gi/gh already contain x@Wih^T+bih and h@Whh^T+bhh.
__global__ __launch_bounds__(256)
void gru_update_kernel(const float* __restrict__ gi, const float* __restrict__ gh,
                       float* __restrict__ hf, bf16* __restrict__ hb)
{
    const long i = (long)blockIdx.x * 256 + threadIdx.x;  // B*HID total
    const long b = i / HID_, j = i % HID_;
    const long base = b * 3 * HID_;
    const float r  = 1.0f / (1.0f + __expf(-(gi[base + j] + gh[base + j])));
    const float z  = 1.0f / (1.0f + __expf(-(gi[base + HID_ + j] + gh[base + HID_ + j])));
    const float n  = tanhf(gi[base + 2 * HID_ + j] + r * gh[base + 2 * HID_ + j]);
    const float h  = (1.0f - z) * n + z * hf[i];
    hf[i] = h;
    hb[i] = (bf16)h;
}

// ---------------------------------------------------------------------------
// Host-side helpers
// ---------------------------------------------------------------------------
static void run_gemm(hipStream_t s, int act, int accum,
                     const bf16* A, int lda, const bf16* W, int ldw,
                     const float* bias, float* Cf, int ldc, bf16* Cb, int ldcb,
                     int M, int N, int K)
{
    dim3 g(N / BN, M / BM), blk(128);
    if (accum)      gemm_bf16_wmma<1, 1><<<g, blk, 0, s>>>(A, lda, W, ldw, bias, Cf, ldc, Cb, ldcb, K);
    else if (act)   gemm_bf16_wmma<1, 0><<<g, blk, 0, s>>>(A, lda, W, ldw, bias, Cf, ldc, Cb, ldcb, K);
    else            gemm_bf16_wmma<0, 0><<<g, blk, 0, s>>>(A, lda, W, ldw, bias, Cf, ldc, Cb, ldcb, K);
}

static void convert(hipStream_t s, const float* in, bf16* out, long n)
{
    f32_to_bf16_kernel<<<dim3((unsigned)((n + 255) / 256)), dim3(256), 0, s>>>(in, out, n);
}

extern "C" void kernel_launch(void* const* d_in, const int* in_sizes, int n_in,
                              void* d_out, int out_size, void* d_ws, size_t ws_size,
                              hipStream_t stream)
{
    (void)in_sizes; (void)n_in; (void)out_size; (void)ws_size;

    const float* obs   = (const float*)d_in[0];
    const float* eps   = (const float*)d_in[1];
    const float* enc_W0 = (const float*)d_in[2];  const float* enc_b0 = (const float*)d_in[3];
    const float* enc_W1 = (const float*)d_in[4];  const float* enc_b1 = (const float*)d_in[5];
    const float* enc_W2 = (const float*)d_in[6];  const float* enc_b2 = (const float*)d_in[7];
    const float* pri_W0 = (const float*)d_in[8];  const float* pri_b0 = (const float*)d_in[9];
    const float* pri_W1 = (const float*)d_in[10]; const float* pri_b1 = (const float*)d_in[11];
    const float* pri_W2 = (const float*)d_in[12]; const float* pri_b2 = (const float*)d_in[13];
    const float* pos_W0 = (const float*)d_in[14]; const float* pos_b0 = (const float*)d_in[15];
    const float* pos_W1 = (const float*)d_in[16]; const float* pos_b1 = (const float*)d_in[17];
    const float* pos_W2 = (const float*)d_in[18]; const float* pos_b2 = (const float*)d_in[19];
    const float* dec_W0 = (const float*)d_in[20]; const float* dec_b0 = (const float*)d_in[21];
    const float* dec_W1 = (const float*)d_in[22]; const float* dec_b1 = (const float*)d_in[23];
    const float* dec_W2 = (const float*)d_in[24]; const float* dec_b2 = (const float*)d_in[25];
    const float* g_Wih  = (const float*)d_in[26]; const float* g_bih  = (const float*)d_in[27];
    const float* g_Whh  = (const float*)d_in[28]; const float* g_bhh  = (const float*)d_in[29];

    // Output layout: recons [B,T,OBS] | kls [B,T] | mu [B,T,LAT]
    float* recons = (float*)d_out;
    float* kls    = recons + (long)BT_ * OBS_;
    float* mu_out = kls + (long)BT_;

    // --------------------------- workspace carve ---------------------------
    size_t off = 0;
    auto carve = [&](size_t nbytes) -> char* {
        off = (off + 255) & ~(size_t)255;
        char* p = (char*)d_ws + off;
        off += nbytes;
        return p;
    };
    auto cb = [&](long nelem) -> bf16*  { return (bf16*)carve((size_t)nelem * sizeof(bf16)); };
    auto cf = [&](long nelem) -> float* { return (float*)carve((size_t)nelem * sizeof(float)); };

    // bf16 weight copies (stay L2-resident across all 128 steps: ~7 MB total)
    bf16* w_enc0 = cb((long)ENC_ * OBS_);
    bf16* w_enc1 = cb((long)ENC_ * ENC_);
    bf16* w_enc2 = cb((long)ENC_ * ENC_);
    bf16* w_pri0 = cb((long)HID_ * HID_);
    bf16* w_pri1 = cb((long)HID_ * HID_);
    bf16* w_pri2 = cb((long)GP_ * HID_);
    bf16* w_pos0 = cb((long)HID_ * (HID_ + ENC_));
    bf16* w_pos1 = cb((long)HID_ * HID_);
    bf16* w_pos2 = cb((long)GP_ * HID_);
    bf16* w_dec0 = cb((long)HID_ * LAT_);
    bf16* w_dec1 = cb((long)HID_ * HID_);
    bf16* w_dec2 = cb((long)OBS_ * HID_);
    bf16* w_wih  = cb((long)3 * HID_ * LAT_);
    bf16* w_whh  = cb((long)3 * HID_ * HID_);

    bf16* obs_bf = cb((long)BT_ * OBS_);   // bf16 obs tokens
    bf16* bufA   = cb((long)BT_ * ENC_);   // enc layer0 out, then enc_all (final)
    bf16* bufB   = cb((long)BT_ * ENC_);   // enc layer1 out

    float* h_f  = cf((long)B_ * HID_);
    bf16*  h_b  = cb((long)B_ * HID_);
    bf16*  t0_b = cb((long)B_ * HID_);
    bf16*  t1_b = cb((long)B_ * HID_);
    float* posp = cf((long)B_ * HID_);     // posterior layer0 split-K partial (f32)
    float* pri_f = cf((long)B_ * GP_);
    float* pos_f = cf((long)B_ * GP_);
    bf16*  z_b  = cb((long)B_ * LAT_);
    float* gi_f = cf((long)B_ * 3 * HID_);
    float* gh_f = cf((long)B_ * 3 * HID_);

    // ---------------------------- setup phase ------------------------------
    convert(stream, enc_W0, w_enc0, (long)ENC_ * OBS_);
    convert(stream, enc_W1, w_enc1, (long)ENC_ * ENC_);
    convert(stream, enc_W2, w_enc2, (long)ENC_ * ENC_);
    convert(stream, pri_W0, w_pri0, (long)HID_ * HID_);
    convert(stream, pri_W1, w_pri1, (long)HID_ * HID_);
    convert(stream, pri_W2, w_pri2, (long)GP_ * HID_);
    convert(stream, pos_W0, w_pos0, (long)HID_ * (HID_ + ENC_));
    convert(stream, pos_W1, w_pos1, (long)HID_ * HID_);
    convert(stream, pos_W2, w_pos2, (long)GP_ * HID_);
    convert(stream, dec_W0, w_dec0, (long)HID_ * LAT_);
    convert(stream, dec_W1, w_dec1, (long)HID_ * HID_);
    convert(stream, dec_W2, w_dec2, (long)OBS_ * HID_);
    convert(stream, g_Wih,  w_wih,  (long)3 * HID_ * LAT_);
    convert(stream, g_Whh,  w_whh,  (long)3 * HID_ * HID_);
    convert(stream, obs,    obs_bf, (long)BT_ * OBS_);

    init_state_kernel<<<dim3((B_ * HID_) / 256), dim3(256), 0, stream>>>(h_f, h_b, (long)B_ * HID_);

    // ------------------- encoder hoisted out of time loop ------------------
    // M = B*T = 131072 tokens, fully parallel (~172 GFLOP of bf16 WMMA GEMM).
    run_gemm(stream, 1, 0, obs_bf, OBS_, w_enc0, OBS_, enc_b0, nullptr, 0, bufA, ENC_, BT_, ENC_, OBS_);
    run_gemm(stream, 1, 0, bufA,   ENC_, w_enc1, ENC_, enc_b1, nullptr, 0, bufB, ENC_, BT_, ENC_, ENC_);
    run_gemm(stream, 0, 0, bufB,   ENC_, w_enc2, ENC_, enc_b2, nullptr, 0, bufA, ENC_, BT_, ENC_, ENC_);
    // bufA now holds enc_all, row (b*T + t).

    // --------------------------- recurrent loop ----------------------------
    for (int t = 0; t < T_; ++t) {
        const bf16* enc_t = bufA + (long)t * ENC_;   // lda = T*ENC picks batch rows

        // prior = mlp3(h)
        run_gemm(stream, 1, 0, h_b,  HID_, w_pri0, HID_, pri_b0, nullptr, 0, t0_b, HID_, B_, HID_, HID_);
        run_gemm(stream, 1, 0, t0_b, HID_, w_pri1, HID_, pri_b1, nullptr, 0, t1_b, HID_, B_, HID_, HID_);
        run_gemm(stream, 0, 0, t1_b, HID_, w_pri2, HID_, pri_b2, pri_f, GP_, nullptr, 0, B_, GP_, HID_);

        // posterior = mlp3(concat(h, enc)): layer0 split into two K=512 GEMMs
        run_gemm(stream, 0, 0, h_b, HID_, w_pos0, HID_ + ENC_, nullptr, posp, HID_, nullptr, 0, B_, HID_, HID_);
        run_gemm(stream, 1, 1, enc_t, T_ * ENC_, w_pos0 + HID_, HID_ + ENC_, pos_b0, posp, HID_, t0_b, HID_, B_, HID_, ENC_);
        run_gemm(stream, 1, 0, t0_b, HID_, w_pos1, HID_, pos_b1, nullptr, 0, t1_b, HID_, B_, HID_, HID_);
        run_gemm(stream, 0, 0, t1_b, HID_, w_pos2, HID_, pos_b2, pos_f, GP_, nullptr, 0, B_, GP_, HID_);

        // reparameterize + KL + mu output
        latent_step_kernel<<<dim3(B_), dim3(64), 0, stream>>>(pri_f, pos_f, eps, z_b, mu_out, kls, t);

        // decoder = mlp3(z) -> recons (strided f32 store straight into d_out)
        run_gemm(stream, 1, 0, z_b,  LAT_, w_dec0, LAT_, dec_b0, nullptr, 0, t0_b, HID_, B_, HID_, LAT_);
        run_gemm(stream, 1, 0, t0_b, HID_, w_dec1, HID_, dec_b1, nullptr, 0, t1_b, HID_, B_, HID_, HID_);
        run_gemm(stream, 0, 0, t1_b, HID_, w_dec2, HID_, dec_b2,
                 recons + (long)t * OBS_, T_ * OBS_, nullptr, 0, B_, OBS_, HID_);

        // GRU: gi = z@Wih^T + bih ; gh = h@Whh^T + bhh ; gate fusion
        run_gemm(stream, 0, 0, z_b, LAT_, w_wih, LAT_, g_bih, gi_f, 3 * HID_, nullptr, 0, B_, 3 * HID_, LAT_);
        run_gemm(stream, 0, 0, h_b, HID_, w_whh, HID_, g_bhh, gh_f, 3 * HID_, nullptr, 0, B_, 3 * HID_, HID_);
        gru_update_kernel<<<dim3((B_ * HID_) / 256), dim3(256), 0, stream>>>(gi_f, gh_f, h_f, h_b);
    }
}
